// AtomAttentionEncoder_74852690035391
// MI455X (gfx1250) — matrix operand order repaired and checked by reading, not compiled, not run on hardware
//
#include <hip/hip_runtime.h>
#include <math.h>

// ---------------------------------------------------------------------------
// AtomAttentionEncoder for MI455X (gfx1250), wave32 + WMMA f32_16x16x32_f16.
//
// Input order assumption: 8 top-level tensors from setup_inputs(), then the
// 'params' dict leaves in insertion order (39 leaves) -> n_in == 47.
//
// Outputs (concatenated f32 in d_out): token_act[T,CT], a[N,CA], q_cond[N,CA],
// pair[S,QW,KW,CP].
// ---------------------------------------------------------------------------

#define DI __device__ __forceinline__

typedef __attribute__((ext_vector_type(16))) _Float16 v16h;
typedef __attribute__((ext_vector_type(8)))  float    v8f;

#define Tc   2048
#define Ac   8
#define CAc  128
#define CPc  16
#define CTc  384
#define NNc  (Tc*Ac)      // 16384
#define QWc  32
#define KWc  128
#define SSc  (NNc/QWc)    // 512
#define NHc  4
#define CHc  32
#define Lc   3
#define HIDc 256

DI v8f wmma_f16(v16h a, v16h b, v8f c) {
  return __builtin_amdgcn_wmma_f32_16x16x32_f16(false, a, false, b, (short)0, c,
                                                false, false);
}

DI float sigmoidf(float x) { return 1.f / (1.f + __expf(-x)); }

// A fragment (16x32, f16) from row-major storage, row stride ld (elements).
// Lane layout: lanes<16 hold K {0..15}, lanes>=16 hold
// K {16..31}, M = lane&15.
DI v16h load_a_rm(const _Float16* p, int ld, int lane) {
  const _Float16* r = p + (size_t)(lane & 15) * ld + ((lane >> 4) << 4);
  v16h a;
#pragma unroll
  for (int j = 0; j < 8; ++j) a[j] = r[j];
#pragma unroll
  for (int j = 0; j < 8; ++j) a[j + 8] = r[8 + j];
  return a;
}

// B fragment (32x16) from TRANSPOSED f16 storage bt[n][k], row stride ld.
// Lane n = lane&15, K half selected by lane>>4 -> 16 contiguous k values.
DI v16h load_bt_rm(const _Float16* bt, int ld, int n0, int k0, int lane) {
  const _Float16* r = bt + (size_t)(n0 + (lane & 15)) * ld + k0 + ((lane >> 4) << 4);
  v16h b;
#pragma unroll
  for (int j = 0; j < 16; ++j) b[j] = r[j];
  return b;
}

// B fragment from row-major fp32 weight W[k][n] (ld = N stride), cvt to f16.
DI v16h load_b_w(const float* W, int ldn, int k0, int n0, int lane) {
  int n = n0 + (lane & 15);
  int k = k0 + ((lane >> 4) << 4);
  v16h b;
#pragma unroll
  for (int j = 0; j < 16; ++j) b[j] = (_Float16)W[(size_t)(k + j) * ldn + n];
  return b;
}

// Same, with per-K diagonal fold (adaLN sln scale folded into the weight).
DI v16h load_b_w_fold(const float* W, int ldn, int k0, int n0,
                      const float* fold, int lane) {
  int n = n0 + (lane & 15);
  int k = k0 + ((lane >> 4) << 4);
  v16h b;
#pragma unroll
  for (int j = 0; j < 16; ++j)
    b[j] = (_Float16)(W[(size_t)(k + j) * ldn + n] * fold[k + j]);
  return b;
}

// 16x16 fp32 weight as a K=32 B fragment with zero padding of rows 16..31.
DI v16h load_b_pad16(const float* W16, int lane) {
  v16h b;
  int n = lane & 15;
  if ((lane >> 4) == 0) {
#pragma unroll
    for (int j = 0; j < 16; ++j) b[j] = (_Float16)W16[j * 16 + n];
  } else {
#pragma unroll
    for (int j = 0; j < 16; ++j) b[j] = (_Float16)0.f;
  }
  return b;
}

// Row LayerNorm (128 ch) f32 -> f16, eps 1e-5.
DI void ln_row_f16(const float* src, _Float16* dst) {
  float mu = 0.f;
#pragma unroll 8
  for (int k = 0; k < CAc; ++k) mu += src[k];
  mu *= (1.f / CAc);
  float var = 0.f;
#pragma unroll 8
  for (int k = 0; k < CAc; ++k) { float d = src[k] - mu; var += d * d; }
  var *= (1.f / CAc);
  float inv = rsqrtf(var + 1e-5f);
#pragma unroll 8
  for (int k = 0; k < CAc; ++k) dst[k] = (_Float16)((src[k] - mu) * inv);
}

// ---------------------------------------------------------------------------
// Kernel 1: per-atom conditioning embed + row-LN cache + q_cond output.
// ---------------------------------------------------------------------------
__global__ __launch_bounds__(128) void k_embed(
    const float* ref_pos, const float* ref_mask, const float* ref_charge,
    const int* ref_element, const int* name_chars,
    const float* W_pos, const float* W_mask, const float* W_el,
    const float* W_chg, const float* W_name,
    float* act, _Float16* ln_act, float* out_q) {
  int n = blockIdx.x;
  int c = threadIdx.x;
  float p0 = ref_pos[n * 3 + 0], p1 = ref_pos[n * 3 + 1], p2 = ref_pos[n * 3 + 2];
  float mk = ref_mask[n];
  float v = p0 * W_pos[c] + p1 * W_pos[CAc + c] + p2 * W_pos[2 * CAc + c];
  v += mk * W_mask[c];
  v += W_el[(size_t)ref_element[n] * CAc + c];
  v += asinhf(ref_charge[n]) * W_chg[c];
#pragma unroll
  for (int j = 0; j < 4; ++j) {
    int ch = name_chars[n * 4 + j];
    v += W_name[(size_t)(j * 64 + ch) * CAc + c];
  }
  v *= mk;
  act[(size_t)n * CAc + c] = v;
  out_q[(size_t)n * CAc + c] = v;

  __shared__ float s1[CAc], s2[CAc];
  s1[c] = v; s2[c] = v * v;
  __syncthreads();
  for (int off = 64; off > 0; off >>= 1) {
    if (c < off) { s1[c] += s1[c + off]; s2[c] += s2[c + off]; }
    __syncthreads();
  }
  float mu = s1[0] * (1.f / CAc);
  float var = s2[0] * (1.f / CAc) - mu * mu;
  ln_act[(size_t)n * CAc + c] = (_Float16)((v - mu) * rsqrtf(var + 1e-5f));
}

// ---------------------------------------------------------------------------
// Kernel 2: pair features per window: init + 3-layer 16x16 WMMA MLP (+skip),
// writes f32 pair output and f16 LN(pair) cache for the attention bias.
// ---------------------------------------------------------------------------
__global__ __launch_bounds__(256) void k_pair(
    const float* act, const int* keys_idx, const float* keys_mask,
    const float* ref_pos, const int* uid,
    const float* W_row, const float* W_col, const float* W_off,
    const float* W_dist, const float* W_pmask, const float* W_mlp,
    float* out_pair, _Float16* pn) {
  __shared__ _Float16 qrel[QWc * CAc];        // relu(q_cond)  8 KB
  __shared__ _Float16 krel[KWc * CAc];        // relu(k_cond) 32 KB
  __shared__ float rowb[QWc * CPc];           //  2 KB
  __shared__ float colb[KWc * CPc];           //  8 KB
  __shared__ float posq[QWc][4];
  __shared__ float posk[KWc][4];
  __shared__ int uidq[QWc];
  __shared__ int uidk[KWc];
  __shared__ _Float16 wscr[8 * 16 * 32];      // per-wave bounce tile, 8 KB

  int s = blockIdx.x;
  int tid = threadIdx.x;
  int wv = tid >> 5, lane = tid & 31;

  for (int e = tid; e < QWc * CAc; e += 256) {
    float v = act[((size_t)s * QWc + (e >> 7)) * CAc + (e & 127)];
    qrel[e] = (_Float16)fmaxf(v, 0.f);
  }
  for (int e = tid; e < KWc * CAc; e += 256) {
    int k = e >> 7;
    int idx = keys_idx[s * KWc + k];
    float m = keys_mask[s * KWc + k];
    float v = (m > 0.f) ? act[(size_t)idx * CAc + (e & 127)] : 0.f;
    krel[e] = (_Float16)fmaxf(v, 0.f);
  }
  if (tid < QWc) {
    int r = s * QWc + tid;
    posq[tid][0] = ref_pos[r * 3 + 0];
    posq[tid][1] = ref_pos[r * 3 + 1];
    posq[tid][2] = ref_pos[r * 3 + 2];
    uidq[tid] = uid[r];
  }
  if (tid < KWc) {
    int idx = keys_idx[s * KWc + tid];
    float m = keys_mask[s * KWc + tid];
    posk[tid][0] = (m > 0.f) ? ref_pos[idx * 3 + 0] : 0.f;
    posk[tid][1] = (m > 0.f) ? ref_pos[idx * 3 + 1] : 0.f;
    posk[tid][2] = (m > 0.f) ? ref_pos[idx * 3 + 2] : 0.f;
    uidk[tid] = (m > 0.f) ? uid[idx] : -1;
  }
  __syncthreads();

  // row = relu(q)@W_row, col = relu(k)@W_col   (K=128, N=16)
  for (int job = wv; job < 10; job += 8) {
    const _Float16* A; const float* W; float* Dst; int mt;
    if (job < 8) { A = krel; W = W_col; Dst = colb; mt = job; }
    else         { A = qrel; W = W_row; Dst = rowb; mt = job - 8; }
    v8f c = {};
    for (int ks = 0; ks < 4; ++ks) {
      v16h a = load_a_rm(A + (size_t)(mt * 16) * CAc + ks * 32, CAc, lane);
      v16h b = load_b_w(W, CPc, ks * 32, 0, lane);
      c = wmma_f16(a, b, c);
    }
    int n = lane & 15, m0 = (lane >> 4) * 8;
#pragma unroll
    for (int j = 0; j < 8; ++j) Dst[(mt * 16 + m0 + j) * CPc + n] = c[j];
  }
  __syncthreads();

  // 256 tiles of 16 (q,k)-rows x 16 channels; each wave 32 tiles.
  _Float16* ta = wscr + wv * 16 * 32;
  for (int tt = wv; tt < 256; tt += 8) {
    int q = tt >> 3, kb = (tt & 7) * 16;
    int cch = lane & 15, m0 = (lane >> 4) * 8;
    float w0 = W_off[cch], w1 = W_off[CPc + cch], w2 = W_off[2 * CPc + cch];
    float wd = W_dist[cch], wp = W_pmask[cch];
    float rv = rowb[q * CPc + cch];
    float qx = posq[q][0], qy = posq[q][1], qz = posq[q][2];
    int qu = uidq[q];
    v8f p0;
#pragma unroll
    for (int j = 0; j < 8; ++j) {
      int k = kb + m0 + j;
      float val = rv + colb[k * CPc + cch];
      if (qu == uidk[k]) {
        float dx = qx - posk[k][0], dy = qy - posk[k][1], dz = qz - posk[k][2];
        float sq = dx * dx + dy * dy + dz * dz;
        val += dx * w0 + dy * w1 + dz * w2 + wd / (1.f + sq) + wp;
      }
      p0[j] = val;
    }
    v8f x = p0;
    for (int l = 0; l < 3; ++l) {
#pragma unroll
      for (int j = 0; j < 8; ++j) {
        ta[(m0 + j) * 32 + cch] = (_Float16)fmaxf(x[j], 0.f);
        ta[(m0 + j) * 32 + cch + 16] = (_Float16)0.f;   // K pad 16..31
      }
      __syncthreads();
      v16h a = load_a_rm(ta, 32, lane);
      v16h b = load_b_pad16(W_mlp + l * CPc * CPc, lane);
      v8f cc = {};
      x = wmma_f16(a, b, cc);
      __syncthreads();
    }
    size_t rbase = (size_t)s * QWc * KWc + (size_t)tt * 16;
    float* tc = (float*)ta;   // reuse as 16x16 f32 bounce for the LN
#pragma unroll
    for (int j = 0; j < 8; ++j) {
      float pf = p0[j] + x[j];
      out_pair[(rbase + m0 + j) * CPc + cch] = pf;
      tc[(m0 + j) * CPc + cch] = pf;
    }
    __syncthreads();
    if (lane < 16) {
      float mu = 0.f;
#pragma unroll
      for (int k = 0; k < 16; ++k) mu += tc[lane * 16 + k];
      mu *= (1.f / 16.f);
      float var = 0.f;
#pragma unroll
      for (int k = 0; k < 16; ++k) { float d = tc[lane * 16 + k] - mu; var += d * d; }
      var *= (1.f / 16.f);
      float inv = rsqrtf(var + 1e-5f);
      _Float16* dst = pn + (rbase + lane) * CPc;
#pragma unroll
      for (int k = 0; k < 16; ++k)
        dst[k] = (_Float16)((tc[lane * 16 + k] - mu) * inv);
    }
    __syncthreads();
  }
}

// ---------------------------------------------------------------------------
// Kernel 3 (per layer): per-atom adaLN (q & k streams) + Q/K/V/gates.
// De-duplicates the gathered key path: computed once per atom, not per key.
// ---------------------------------------------------------------------------
__global__ __launch_bounds__(256) void k_qkvprep(
    const float* a_in, const _Float16* ln_act, const float* act,
    const float* aq_sln, const float* aq_Wg, const float* aq_bg, const float* aq_Wb,
    const float* ak_sln, const float* ak_Wg, const float* ak_bg, const float* ak_Wb,
    const float* Wq, const float* bq, const float* Wk, const float* Wv, const float* Wgt,
    const float* o_Wgs, const float* o_bgs, const float* t_gWs, const float* t_gbs,
    _Float16* Qh, _Float16* Kh, _Float16* Vh, _Float16* Gg, _Float16* OGS,
    _Float16* TG) {
  __shared__ _Float16 lns[32 * CAc];   // LN(cond) (then raw cond, phase 2)
  __shared__ _Float16 lna[32 * CAc];   // LN(a)
  __shared__ _Float16 aq[32 * CAc];
  __shared__ _Float16 ak[32 * CAc];
  int row0 = blockIdx.x * 32;
  int tid = threadIdx.x, wv = tid >> 5, lane = tid & 31;
  const float inv_sqrt_c = 0.17677669529663687f;   // 1/sqrt(CH)

  for (int e = tid; e < 32 * CAc; e += 256) lns[e] = ln_act[(size_t)row0 * CAc + e];
  if (tid < 32) ln_row_f16(a_in + (size_t)(row0 + tid) * CAc, lna + tid * CAc);
  __syncthreads();

  for (int job = wv; job < 16; job += 8) {
    int mt = job >> 3, nt = job & 7;
    v8f gq = {}, bq2 = {}, gk = {}, bk2 = {};
    for (int ks = 0; ks < 4; ++ks) {
      v16h a = load_a_rm(lns + (size_t)(mt * 16) * CAc + ks * 32, CAc, lane);
      gq  = wmma_f16(a, load_b_w_fold(aq_Wg, CAc, ks * 32, nt * 16, aq_sln, lane), gq);
      bq2 = wmma_f16(a, load_b_w_fold(aq_Wb, CAc, ks * 32, nt * 16, aq_sln, lane), bq2);
      gk  = wmma_f16(a, load_b_w_fold(ak_Wg, CAc, ks * 32, nt * 16, ak_sln, lane), gk);
      bk2 = wmma_f16(a, load_b_w_fold(ak_Wb, CAc, ks * 32, nt * 16, ak_sln, lane), bk2);
    }
    int n = nt * 16 + (lane & 15), m0 = (lane >> 4) * 8;
    float bgq = aq_bg[n], bgk = ak_bg[n];
#pragma unroll
    for (int j = 0; j < 8; ++j) {
      int r = mt * 16 + m0 + j;
      float l = (float)lna[r * CAc + n];
      aq[r * CAc + n] = (_Float16)(sigmoidf(gq[j] + bgq) * l + bq2[j]);
      ak[r * CAc + n] = (_Float16)(sigmoidf(gk[j] + bgk) * l + bk2[j]);
    }
  }
  __syncthreads();

  for (int job = wv; job < 16; job += 8) {
    int mt = job >> 3, nt = job & 7;
    v8f cq = {}, ck = {}, cv = {}, cg = {};
    for (int ks = 0; ks < 4; ++ks) {
      v16h aa = load_a_rm(aq + (size_t)(mt * 16) * CAc + ks * 32, CAc, lane);
      v16h ab = load_a_rm(ak + (size_t)(mt * 16) * CAc + ks * 32, CAc, lane);
      cq = wmma_f16(aa, load_b_w(Wq,  CAc, ks * 32, nt * 16, lane), cq);
      cg = wmma_f16(aa, load_b_w(Wgt, CAc, ks * 32, nt * 16, lane), cg);
      ck = wmma_f16(ab, load_b_w(Wk,  CAc, ks * 32, nt * 16, lane), ck);
      cv = wmma_f16(ab, load_b_w(Wv,  CAc, ks * 32, nt * 16, lane), cv);
    }
    int n = nt * 16 + (lane & 15), m0 = (lane >> 4) * 8;
    float bqn = bq[n];
#pragma unroll
    for (int j = 0; j < 8; ++j) {
      size_t r = (size_t)(row0 + mt * 16 + m0 + j) * CAc + n;
      Qh[r] = (_Float16)((cq[j] + bqn) * inv_sqrt_c);
      Kh[r] = (_Float16)ck[j];
      Vh[r] = (_Float16)cv[j];
      Gg[r] = (_Float16)sigmoidf(cg[j]);
    }
  }
  __syncthreads();

  for (int e = tid; e < 32 * CAc; e += 256)
    lns[e] = (_Float16)act[(size_t)row0 * CAc + e];   // raw cond for gates
  __syncthreads();
  for (int job = wv; job < 16; job += 8) {
    int mt = job >> 3, nt = job & 7;
    v8f c1 = {}, c2 = {};
    for (int ks = 0; ks < 4; ++ks) {
      v16h a = load_a_rm(lns + (size_t)(mt * 16) * CAc + ks * 32, CAc, lane);
      c1 = wmma_f16(a, load_b_w(o_Wgs, CAc, ks * 32, nt * 16, lane), c1);
      c2 = wmma_f16(a, load_b_w(t_gWs, CAc, ks * 32, nt * 16, lane), c2);
    }
    int n = nt * 16 + (lane & 15), m0 = (lane >> 4) * 8;
    float b1 = o_bgs[n], b2 = t_gbs[n];
#pragma unroll
    for (int j = 0; j < 8; ++j) {
      size_t r = (size_t)(row0 + mt * 16 + m0 + j) * CAc + n;
      OGS[r] = (_Float16)sigmoidf(c1[j] + b1);
      TG[r]  = (_Float16)sigmoidf(c2[j] + b2);
    }
  }
}

// ---------------------------------------------------------------------------
// Kernel 4 (per layer): windowed pair-biased attention, fully fused in LDS.
// ---------------------------------------------------------------------------
__global__ __launch_bounds__(256) void k_attn(
    const float* a_in, const _Float16* Qh, const _Float16* Kh, const _Float16* Vh,
    const _Float16* Gg, const _Float16* OGS, const _Float16* pn,
    const int* keys_idx, const float* keys_mask,
    const float* z_sln, const float* Wbias, const float* Wout, float* a_mid) {
  __shared__ _Float16 bufA[KWc * CAc];        // Kg then Vt     (32 KB)
  __shared__ _Float16 bufB[NHc * QWc * KWc];  // scores then o  (32 KB)
  int s = blockIdx.x, tid = threadIdx.x, wv = tid >> 5, lane = tid & 31;

  for (int e = tid; e < KWc * CAc; e += 256) {     // gather K (km-zeroed)
    int k = e >> 7, ch = e & 127;
    int idx = keys_idx[s * KWc + k];
    float m = keys_mask[s * KWc + k];
    bufA[e] = (m > 0.f) ? Kh[(size_t)idx * CAc + ch] : (_Float16)0.f;
  }
  __syncthreads();

  // scores: per head CH=32 == one WMMA K-step
  for (int id = wv; id < 64; id += 8) {
    int h = id >> 4, mt = (id >> 3) & 1, nt = id & 7;
    v16h a = load_a_rm(Qh + (size_t)(s * QWc + mt * 16) * CAc + h * CHc, CAc, lane);
    v16h b = load_bt_rm(bufA, CAc, nt * 16, h * CHc, lane);
    v8f c = {};
    c = wmma_f16(a, b, c);
    int key = nt * 16 + (lane & 15), m0 = (lane >> 4) * 8;
    float wb[16];
#pragma unroll
    for (int cc = 0; cc < 16; ++cc) wb[cc] = z_sln[cc] * Wbias[cc * NHc + h];
#pragma unroll
    for (int j = 0; j < 8; ++j) {
      int q = mt * 16 + m0 + j;
      const _Float16* pp = pn + ((size_t)(s * QWc + q) * KWc + key) * CPc;
      float bias = 0.f;
#pragma unroll
      for (int cc = 0; cc < 16; ++cc) bias += (float)pp[cc] * wb[cc];
      bufB[(size_t)(h * QWc + q) * KWc + key] = (_Float16)(c[j] + bias);
    }
  }
  __syncthreads();

  if (tid < NHc * QWc) {                           // softmax over 128 keys
    _Float16* row = bufB + (size_t)tid * KWc;
    float mx = -1e30f;
    for (int k = 0; k < KWc; ++k) mx = fmaxf(mx, (float)row[k]);
    float sm = 0.f;
    for (int k = 0; k < KWc; ++k) sm += __expf((float)row[k] - mx);
    float inv = 1.f / sm;
    for (int k = 0; k < KWc; ++k)
      row[k] = (_Float16)(__expf((float)row[k] - mx) * inv);
  }
  __syncthreads();

  for (int e = tid; e < KWc * CAc; e += 256) {     // gather V transposed
    int k = e >> 7, ch = e & 127;
    int idx = keys_idx[s * KWc + k];
    float m = keys_mask[s * KWc + k];
    bufA[(size_t)ch * KWc + k] = (m > 0.f) ? Vh[(size_t)idx * CAc + ch]
                                           : (_Float16)0.f;
  }
  __syncthreads();

  v8f oo[2]; int oid[2]; int cnt = 0;              // o = att@V, gated by G
  for (int id = wv; id < 16; id += 8) {
    int h = id >> 2, mt = (id >> 1) & 1, ntc = id & 1;
    v8f o = {};
    for (int ks = 0; ks < 4; ++ks) {
      v16h a = load_a_rm(bufB + (size_t)(h * QWc + mt * 16) * KWc + ks * 32, KWc, lane);
      v16h b = load_bt_rm(bufA, KWc, h * CHc + ntc * 16, ks * 32, lane);
      o = wmma_f16(a, b, o);
    }
    int n = lane & 15, m0 = (lane >> 4) * 8;
    int ch = h * CHc + ntc * 16 + n;
#pragma unroll
    for (int j = 0; j < 8; ++j) {
      size_t r = (size_t)(s * QWc + mt * 16 + m0 + j);
      o[j] *= (float)Gg[r * CAc + ch];
    }
    oo[cnt] = o; oid[cnt] = id; ++cnt;
  }
  __syncthreads();
  for (int t = 0; t < cnt; ++t) {                  // bounce o as f16 A-operand
    int id = oid[t]; int h = id >> 2, mt = (id >> 1) & 1, ntc = id & 1;
    int n = lane & 15, m0 = (lane >> 4) * 8;
    int ch = h * CHc + ntc * 16 + n;
#pragma unroll
    for (int j = 0; j < 8; ++j)
      bufB[(size_t)(mt * 16 + m0 + j) * CAc + ch] = (_Float16)oo[t][j];
  }
  __syncthreads();

  for (int id = wv; id < 16; id += 8) {            // (o@Wout)*OGS + residual
    int mt = id >> 3, nt = id & 7;
    v8f c = {};
    for (int ks = 0; ks < 4; ++ks) {
      v16h a = load_a_rm(bufB + (size_t)(mt * 16) * CAc + ks * 32, CAc, lane);
      c = wmma_f16(a, load_b_w(Wout, CAc, ks * 32, nt * 16, lane), c);
    }
    int n = nt * 16 + (lane & 15), m0 = (lane >> 4) * 8;
#pragma unroll
    for (int j = 0; j < 8; ++j) {
      size_t r = (size_t)(s * QWc + mt * 16 + m0 + j) * CAc + n;
      a_mid[r] = a_in[r] + (float)OGS[r] * c[j];
    }
  }
}

// ---------------------------------------------------------------------------
// Kernel 5 (per layer): conditioned transition block (128->256->128, SwiGLU).
// ---------------------------------------------------------------------------
__global__ __launch_bounds__(256) void k_trans(
    const float* a_mid, const _Float16* ln_act,
    const float* t_sln, const float* t_Wg, const float* t_bg, const float* t_Wb,
    const float* t_Wsw, const float* t_Wlin, const float* t_Wo,
    const _Float16* TG, float* a_out) {
  __shared__ _Float16 lns[32 * CAc];
  __shared__ _Float16 lt[32 * CAc];      // LN(a_mid) then t (in place)
  __shared__ _Float16 hid[32 * HIDc];
  int row0 = blockIdx.x * 32, tid = threadIdx.x, wv = tid >> 5, lane = tid & 31;

  for (int e = tid; e < 32 * CAc; e += 256) lns[e] = ln_act[(size_t)row0 * CAc + e];
  if (tid < 32) ln_row_f16(a_mid + (size_t)(row0 + tid) * CAc, lt + tid * CAc);
  __syncthreads();

  for (int job = wv; job < 16; job += 8) {
    int mt = job >> 3, nt = job & 7;
    v8f g = {}, b2 = {};
    for (int ks = 0; ks < 4; ++ks) {
      v16h a = load_a_rm(lns + (size_t)(mt * 16) * CAc + ks * 32, CAc, lane);
      g  = wmma_f16(a, load_b_w_fold(t_Wg, CAc, ks * 32, nt * 16, t_sln, lane), g);
      b2 = wmma_f16(a, load_b_w_fold(t_Wb, CAc, ks * 32, nt * 16, t_sln, lane), b2);
    }
    int n = nt * 16 + (lane & 15), m0 = (lane >> 4) * 8;
    float bg = t_bg[n];
#pragma unroll
    for (int j = 0; j < 8; ++j) {
      int r = mt * 16 + m0 + j;
      float l = (float)lt[r * CAc + n];
      lt[r * CAc + n] = (_Float16)(sigmoidf(g[j] + bg) * l + b2[j]);
    }
  }
  __syncthreads();

  for (int job = wv; job < 32; job += 8) {
    int mt = job >> 4, nt = job & 15;
    v8f c1 = {}, c2 = {};
    for (int ks = 0; ks < 4; ++ks) {
      v16h a = load_a_rm(lt + (size_t)(mt * 16) * CAc + ks * 32, CAc, lane);
      c1 = wmma_f16(a, load_b_w(t_Wsw,  HIDc, ks * 32, nt * 16, lane), c1);
      c2 = wmma_f16(a, load_b_w(t_Wlin, HIDc, ks * 32, nt * 16, lane), c2);
    }
    int n = nt * 16 + (lane & 15), m0 = (lane >> 4) * 8;
#pragma unroll
    for (int j = 0; j < 8; ++j) {
      int r = mt * 16 + m0 + j;
      float x = c1[j];
      hid[(size_t)r * HIDc + n] = (_Float16)(x * sigmoidf(x) * c2[j]);
    }
  }
  __syncthreads();

  for (int job = wv; job < 16; job += 8) {
    int mt = job >> 3, nt = job & 7;
    v8f c = {};
    for (int ks = 0; ks < 8; ++ks) {
      v16h a = load_a_rm(hid + (size_t)(mt * 16) * HIDc + ks * 32, HIDc, lane);
      c = wmma_f16(a, load_b_w(t_Wo, CAc, ks * 32, nt * 16, lane), c);
    }
    int n = nt * 16 + (lane & 15), m0 = (lane >> 4) * 8;
#pragma unroll
    for (int j = 0; j < 8; ++j) {
      size_t r = (size_t)(row0 + mt * 16 + m0 + j) * CAc + n;
      a_out[r] = a_mid[r] + (float)TG[r] * c[j];
    }
  }
}

// ---------------------------------------------------------------------------
// Kernel 6: mask, a output, token projection (CA->CT WMMA) + mean pool.
// ---------------------------------------------------------------------------
__global__ __launch_bounds__(256) void k_out(
    const float* a_fin, const float* ref_mask, const float* W_proj,
    float* out_a, float* out_tok) {
  __shared__ _Float16 af[32 * CAc];   //  8 KB
  __shared__ _Float16 tk[32 * CTc];   // 24 KB
  int row0 = blockIdx.x * 32, tid = threadIdx.x, wv = tid >> 5, lane = tid & 31;

  for (int e = tid; e < 32 * CAc; e += 256) {
    int r = row0 + (e >> 7);
    float v = a_fin[(size_t)r * CAc + (e & 127)] * ref_mask[r];
    af[e] = (_Float16)v;
    out_a[(size_t)row0 * CAc + e] = v;
  }
  __syncthreads();

  for (int job = wv; job < 48; job += 8) {
    int mt = job / 24, nt = job % 24;
    v8f c = {};
    for (int ks = 0; ks < 4; ++ks) {
      v16h a = load_a_rm(af + (size_t)(mt * 16) * CAc + ks * 32, CAc, lane);
      c = wmma_f16(a, load_b_w(W_proj, CTc, ks * 32, nt * 16, lane), c);
    }
    int n = nt * 16 + (lane & 15), m0 = (lane >> 4) * 8;
#pragma unroll
    for (int j = 0; j < 8; ++j)
      tk[(size_t)(mt * 16 + m0 + j) * CTc + n] = (_Float16)fmaxf(c[j], 0.f);
  }
  __syncthreads();

  for (int e = tid; e < 4 * CTc; e += 256) {       // 4 tokens per block
    int t = e / CTc, ch = e % CTc;
    float sum = 0.f, msum = 0.f;
#pragma unroll
    for (int r = 0; r < 8; ++r) {
      float m = ref_mask[row0 + t * 8 + r];
      sum  += (float)tk[(size_t)(t * 8 + r) * CTc + ch] * m;
      msum += m;
    }
    out_tok[(size_t)(row0 / 8 + t) * CTc + ch] = sum / fmaxf(msum, 1e-10f);
  }
}

// ---------------------------------------------------------------------------
extern "C" void kernel_launch(void* const* d_in, const int* in_sizes, int n_in,
                              void* d_out, int out_size, void* d_ws, size_t ws_size,
                              hipStream_t stream) {
  (void)in_sizes; (void)n_in; (void)out_size; (void)ws_size;

  const float* ref_pos   = (const float*)d_in[0];
  const float* ref_mask  = (const float*)d_in[1];
  const float* ref_chg   = (const float*)d_in[2];
  const int*   ref_el    = (const int*)d_in[3];
  const int*   ref_name  = (const int*)d_in[4];
  const int*   ref_uid   = (const int*)d_in[5];
  const int*   keys_idx  = (const int*)d_in[6];
  const float* keys_mask = (const float*)d_in[7];

  // params leaves in setup_inputs() insertion order
  enum { pW_ref_pos = 0, pW_ref_mask, pW_ref_element, pW_ref_charge, pW_ref_name,
         pW_row, pW_col, pW_off, pW_dist, pW_pmask, pW_mlp, pW_proj,
         paq_sln, paq_Wg, paq_bg, paq_Wb, pak_sln, pak_Wg, pak_bg, pak_Wb, pz_sln,
         pWq, pbq, pWk, pWv, pWbias, pWg, pWout, po_Wgs, po_bgs,
         pt_sln, pt_Wg, pt_bg, pt_Wb, pt_Wsw, pt_Wlin, pt_Wo, pt_gWs, pt_gbs,
         P_COUNT };
  const float* P[P_COUNT];
  for (int i = 0; i < P_COUNT; ++i) P[i] = (const float*)d_in[8 + i];

  float* out_tok  = (float*)d_out;
  float* out_a    = out_tok + (size_t)Tc * CTc;
  float* out_q    = out_a + (size_t)NNc * CAc;
  float* out_pair = out_q + (size_t)NNc * CAc;

  // workspace carve-out (~116 MB)
  char* w = (char*)d_ws;
  auto alloc = [&](size_t b) { char* p = w; w += (b + 255) & ~(size_t)255; return p; };
  float*    act    = (float*)alloc((size_t)NNc * CAc * 4);
  _Float16* ln_act = (_Float16*)alloc((size_t)NNc * CAc * 2);
  _Float16* pn     = (_Float16*)alloc((size_t)SSc * QWc * KWc * CPc * 2);
  _Float16* Qh     = (_Float16*)alloc((size_t)NNc * CAc * 2);
  _Float16* Kh     = (_Float16*)alloc((size_t)NNc * CAc * 2);
  _Float16* Vh     = (_Float16*)alloc((size_t)NNc * CAc * 2);
  _Float16* Gg     = (_Float16*)alloc((size_t)NNc * CAc * 2);
  _Float16* OGS    = (_Float16*)alloc((size_t)NNc * CAc * 2);
  _Float16* TG     = (_Float16*)alloc((size_t)NNc * CAc * 2);
  float*    A0     = (float*)alloc((size_t)NNc * CAc * 4);
  float*    A1     = (float*)alloc((size_t)NNc * CAc * 4);

  k_embed<<<NNc, 128, 0, stream>>>(ref_pos, ref_mask, ref_chg, ref_el, ref_name,
      P[pW_ref_pos], P[pW_ref_mask], P[pW_ref_element], P[pW_ref_charge],
      P[pW_ref_name], act, ln_act, out_q);

  k_pair<<<SSc, 256, 0, stream>>>(act, keys_idx, keys_mask, ref_pos, ref_uid,
      P[pW_row], P[pW_col], P[pW_off], P[pW_dist], P[pW_pmask], P[pW_mlp],
      out_pair, pn);

  for (int i = 0; i < Lc; ++i) {
    const float* a_in = (i == 0) ? act : A0;
    size_t o2 = (size_t)i * CAc * CAc, o1 = (size_t)i * CAc;
    k_qkvprep<<<NNc / 32, 256, 0, stream>>>(a_in, ln_act, act,
        P[paq_sln] + o1, P[paq_Wg] + o2, P[paq_bg] + o1, P[paq_Wb] + o2,
        P[pak_sln] + o1, P[pak_Wg] + o2, P[pak_bg] + o1, P[pak_Wb] + o2,
        P[pWq] + o2, P[pbq] + o1, P[pWk] + o2, P[pWv] + o2, P[pWg] + o2,
        P[po_Wgs] + o2, P[po_bgs] + o1, P[pt_gWs] + o2, P[pt_gbs] + o1,
        Qh, Kh, Vh, Gg, OGS, TG);
    k_attn<<<SSc, 256, 0, stream>>>(a_in, Qh, Kh, Vh, Gg, OGS, pn,
        keys_idx, keys_mask,
        P[pz_sln] + (size_t)i * CPc, P[pWbias] + (size_t)i * CPc * NHc,
        P[pWout] + o2, A1);
    k_trans<<<NNc / 32, 256, 0, stream>>>(A1, ln_act,
        P[pt_sln] + o1, P[pt_Wg] + o2, P[pt_bg] + o1, P[pt_Wb] + o2,
        P[pt_Wsw] + (size_t)i * CAc * HIDc, P[pt_Wlin] + (size_t)i * CAc * HIDc,
        P[pt_Wo] + (size_t)i * HIDc * CAc, TG, A0);
  }

  k_out<<<NNc / 32, 256, 0, stream>>>(A0, ref_mask, P[pW_proj], out_a, out_tok);
}
